// GRNDrugGCN_21560735825958
// MI455X (gfx1250) — matrix operand order repaired and checked by reading, not compiled, not run on hardware
//
#include <hip/hip_runtime.h>
#include <hip/hip_bf16.h>

typedef __attribute__((ext_vector_type(16))) _Float16 v16h;
typedef __attribute__((ext_vector_type(8)))  float    v8f;

// ---------------------------------------------------------------------------
// WMMA fragment helpers (wave32, v_wmma_f32_16x16x32_f16)
// A (16x32 f16): lane m = lane&15, half = lane>>4, K(j) = 16*half + j
// B (32x16 f16): lane n = lane&15, half = lane>>4, K(j) = 16*half + j
// C/D (16x16 f32): c[r] = C[r + 8*half][lane&15]
// ---------------------------------------------------------------------------

__device__ __forceinline__ int kmapA(int j, int half) {
    return 16 * half + j;
}

// Load one A-fragment row slice (32 consecutive K floats) and convert to f16.
__device__ __forceinline__ v16h load_rowA(const float* __restrict__ rowp, int half) {
    v16h a;
#pragma unroll
    for (int j = 0; j < 16; j += 2) {
        int k = kmapA(j, half);                      // k and k+1 are contiguous
        float2 p = *reinterpret_cast<const float2*>(rowp + k);
        a[j]     = (_Float16)p.x;
        a[j + 1] = (_Float16)p.y;
    }
    return a;
}

__device__ __forceinline__ v8f wmma_f16(v16h a, v16h b, v8f c) {
    return __builtin_amdgcn_wmma_f32_16x16x32_f16(false, a, false, b,
                                                  (short)0, c, false, false);
}

// ---------------------------------------------------------------------------
// Dense GEMM: out[nrows,32] = A[nrows, KC*32] @ W[KC*32, 32]   (f16 WMMA, f32 acc)
// One wave per 16-row tile; two 16x16 column tiles per wave.
// ---------------------------------------------------------------------------
template <int KC>
__global__ void wmma_gemm(const float* __restrict__ A,
                          const float* __restrict__ W,
                          float* __restrict__ out,
                          int nrows, int ntiles) {
    int wave = threadIdx.x >> 5;
    int tile = blockIdx.x * (blockDim.x >> 5) + wave;
    if (tile >= ntiles) return;                      // wave-uniform exit
    int lane = threadIdx.x & 31;
    int low  = lane & 15;
    int half = lane >> 4;

    // Weight fragments (tiny matrices: load straight from global, cvt to f16)
    v16h Bf[KC][2];
#pragma unroll
    for (int kc = 0; kc < KC; ++kc)
#pragma unroll
        for (int nh = 0; nh < 2; ++nh)
#pragma unroll
            for (int j = 0; j < 16; ++j)
                Bf[kc][nh][j] = (_Float16)W[(kc * 32 + 16 * half + j) * 32 + nh * 16 + low];

    int arow = tile * 16 + low;
    if (arow >= nrows) arow = nrows - 1;             // clamp (loads only)
    const float* rowp = A + (size_t)arow * (KC * 32);

    v8f c0 = {}, c1 = {};
#pragma unroll
    for (int kc = 0; kc < KC; ++kc) {
        v16h a = load_rowA(rowp + kc * 32, half);
        c0 = wmma_f16(a, Bf[kc][0], c0);
        c1 = wmma_f16(a, Bf[kc][1], c1);
    }

#pragma unroll
    for (int r = 0; r < 8; ++r) {
        int orow = tile * 16 + 8 * half + r;
        if (orow < nrows) {
            out[(size_t)orow * 32 + low]      = c0[r];
            out[(size_t)orow * 32 + 16 + low] = c1[r];
        }
    }
}

// ---------------------------------------------------------------------------
// Degree / normalization
// ---------------------------------------------------------------------------
__global__ void deg_count(const int* __restrict__ dst, float* __restrict__ deg, int E) {
    int e = blockIdx.x * blockDim.x + threadIdx.x;
    if (e < E) atomicAdd(deg + dst[e], 1.0f);
}

__global__ void deg_to_dinv(float* __restrict__ deg, int N) {
    int i = blockIdx.x * blockDim.x + threadIdx.x;
    if (i < N) deg[i] = rsqrtf(deg[i] + 1.0f);
}

// ---------------------------------------------------------------------------
// Scatter-add aggregation: one lane per (edge, channel). Gather of hw[src]
// is a coalesced 128B row read; atomics land in L2 (agg = 12.8MB << 192MB L2).
// ---------------------------------------------------------------------------
__global__ void scatter_agg(const int* __restrict__ src, const int* __restrict__ dst,
                            const float* __restrict__ dinv,
                            const float* __restrict__ hw,
                            float* __restrict__ agg, int E) {
    long long g = (long long)blockIdx.x * blockDim.x + threadIdx.x;
    int e = (int)(g >> 5);
    int c = (int)(g & 31);
    if (e >= E) return;
    int s = src[e], d = dst[e];
    float nrm = dinv[s] * dinv[d];
    atomicAdd(agg + (size_t)d * 32 + c, hw[(size_t)s * 32 + c] * nrm);
}

// h = relu(agg + hw * dinv^2 + b)   (in-place into agg)
__global__ void finalize_layer(float* __restrict__ agg, const float* __restrict__ hw,
                               const float* __restrict__ dinv,
                               const float* __restrict__ bias, int N) {
    long long g = (long long)blockIdx.x * blockDim.x + threadIdx.x;
    int i = (int)(g >> 5);
    int c = (int)(g & 31);
    if (i >= N) return;
    float di = dinv[i];
    float v = agg[g] + hw[g] * di * di + bias[c];
    agg[g] = fmaxf(v, 0.0f);
}

// ---------------------------------------------------------------------------
// Gene head: out[row] = relu(h[gid[row]] @ gW1 + gb1) @ gW2 + gb2
// One wave per 16-row tile; WMMA 16x16x32 then shfl-reduce over n.
// ---------------------------------------------------------------------------
__global__ void gene_head(const float* __restrict__ h, const int* __restrict__ gid,
                          const float* __restrict__ gW1, const float* __restrict__ gb1,
                          const float* __restrict__ gW2, const float* __restrict__ gb2,
                          float* __restrict__ out, int nrows, int ntiles) {
    int wave = threadIdx.x >> 5;
    int tile = blockIdx.x * (blockDim.x >> 5) + wave;
    if (tile >= ntiles) return;
    int lane = threadIdx.x & 31;
    int low  = lane & 15;
    int half = lane >> 4;

    v16h B0, B1;
#pragma unroll
    for (int j = 0; j < 16; ++j) {
        int k = 16 * half + j;
        B0[j] = (_Float16)gW1[k * 32 + low];
        B1[j] = (_Float16)gW1[k * 32 + 16 + low];
    }

    int row = tile * 16 + low;
    if (row >= nrows) row = nrows - 1;
    int idx = gid[row];
    v16h a = load_rowA(h + (size_t)idx * 32, half);

    v8f c0 = {}, c1 = {};
    c0 = wmma_f16(a, B0, c0);
    c1 = wmma_f16(a, B1, c1);

    float b1l = gb1[low], b1h = gb1[16 + low];
    float w2l = gW2[low], w2h = gW2[16 + low];
    float b2  = gb2[0];

#pragma unroll
    for (int r = 0; r < 8; ++r) {
        float acc = fmaxf(c0[r] + b1l, 0.0f) * w2l + fmaxf(c1[r] + b1h, 0.0f) * w2h;
#pragma unroll
        for (int m = 1; m < 16; m <<= 1) acc += __shfl_xor(acc, m, 32);
        int orow = tile * 16 + 8 * half + r;
        if (low == 0 && orow < nrows) out[orow] = acc + b2;
    }
}

// ---------------------------------------------------------------------------
// Edge head: concat(h[tf], h[gene], ef[2]) -> K=66 padded to 96 (3 k-chunks)
// out[row] = relu(cat @ eW1 + eb1) @ eW2 + eb2
// ---------------------------------------------------------------------------
__global__ void edge_head(const float* __restrict__ h,
                          const int* __restrict__ tfi, const int* __restrict__ gni,
                          const float* __restrict__ ef,
                          const float* __restrict__ eW1, const float* __restrict__ eb1,
                          const float* __restrict__ eW2, const float* __restrict__ eb2,
                          float* __restrict__ out, int nrows, int ntiles) {
    int wave = threadIdx.x >> 5;
    int tile = blockIdx.x * (blockDim.x >> 5) + wave;
    if (tile >= ntiles) return;
    int lane = threadIdx.x & 31;
    int low  = lane & 15;
    int half = lane >> 4;

    // Weight fragments: chunks 0,1 full; chunk 2 has only rows 64,65 valid
    v16h Bf[3][2];
#pragma unroll
    for (int kc = 0; kc < 3; ++kc)
#pragma unroll
        for (int nh = 0; nh < 2; ++nh)
#pragma unroll
            for (int j = 0; j < 16; ++j) {
                int k = kc * 32 + 16 * half + j;
                float w = (k < 66) ? eW1[k * 32 + nh * 16 + low] : 0.0f;
                Bf[kc][nh][j] = (_Float16)w;
            }

    int row = tile * 16 + low;
    if (row >= nrows) row = nrows - 1;
    int ti = tfi[row];
    int gi = gni[row];
    float ef0 = ef[(size_t)row * 2];
    float ef1 = ef[(size_t)row * 2 + 1];

    v8f c0 = {}, c1 = {};
    {   // chunk 0: h[tf] channels 0..31
        v16h a = load_rowA(h + (size_t)ti * 32, half);
        c0 = wmma_f16(a, Bf[0][0], c0);
        c1 = wmma_f16(a, Bf[0][1], c1);
    }
    {   // chunk 1: h[gene] channels 0..31
        v16h a = load_rowA(h + (size_t)gi * 32, half);
        c0 = wmma_f16(a, Bf[1][0], c0);
        c1 = wmma_f16(a, Bf[1][1], c1);
    }
    {   // chunk 2: edge features at K=64,65; rest zero pad
        v16h a;
#pragma unroll
        for (int j = 0; j < 16; ++j) {
            int k = 64 + kmapA(j, half);
            float v = (k == 64) ? ef0 : (k == 65 ? ef1 : 0.0f);
            a[j] = (_Float16)v;
        }
        c0 = wmma_f16(a, Bf[2][0], c0);
        c1 = wmma_f16(a, Bf[2][1], c1);
    }

    float b1l = eb1[low], b1h = eb1[16 + low];
    float w2l = eW2[low], w2h = eW2[16 + low];
    float b2  = eb2[0];

#pragma unroll
    for (int r = 0; r < 8; ++r) {
        float acc = fmaxf(c0[r] + b1l, 0.0f) * w2l + fmaxf(c1[r] + b1h, 0.0f) * w2h;
#pragma unroll
        for (int m = 1; m < 16; m <<= 1) acc += __shfl_xor(acc, m, 32);
        int orow = tile * 16 + 8 * half + r;
        if (low == 0 && orow < nrows) out[orow] = acc + b2;
    }
}

// ---------------------------------------------------------------------------
// Launcher
// ---------------------------------------------------------------------------
extern "C" void kernel_launch(void* const* d_in, const int* in_sizes, int n_in,
                              void* d_out, int out_size, void* d_ws, size_t ws_size,
                              hipStream_t stream) {
    (void)n_in; (void)out_size; (void)ws_size;

    const float* x   = (const float*)d_in[0];
    const int*   ei  = (const int*)d_in[1];
    const int*   tfi = (const int*)d_in[2];
    const int*   gni = (const int*)d_in[3];
    const float* ef  = (const float*)d_in[4];
    const int*   gid = (const int*)d_in[5];
    const float* W1  = (const float*)d_in[6];
    const float* b1  = (const float*)d_in[7];
    const float* W2  = (const float*)d_in[8];
    const float* b2  = (const float*)d_in[9];
    const float* eW1 = (const float*)d_in[10];
    const float* eb1 = (const float*)d_in[11];
    const float* eW2 = (const float*)d_in[12];
    const float* eb2 = (const float*)d_in[13];
    const float* gW1 = (const float*)d_in[14];
    const float* gb1 = (const float*)d_in[15];
    const float* gW2 = (const float*)d_in[16];
    const float* gb2 = (const float*)d_in[17];

    const int N  = in_sizes[0] / 128;   // 100000
    const int E  = in_sizes[1] / 2;     // 3200000
    const int EP = in_sizes[2];         // 1000000
    const int NG = in_sizes[5];         // 50000
    const int* src = ei;
    const int* dst = ei + E;

    float* ws   = (float*)d_ws;
    float* dinv = ws;                         // N floats
    float* hw   = ws + N;                     // N*32 floats (hw1 / hw2)
    float* agg  = ws + N + (size_t)N * 32;    // N*32 floats (agg / h1 / h2)
    float* outp = (float*)d_out;

    const int THREADS = 256;
    const int ntilesN = (N  + 15) / 16;
    const int ntilesE = (EP + 15) / 16;
    const int ntilesG = (NG + 15) / 16;
    const long long NC = (long long)N * 32;
    const long long EC = (long long)E * 32;

    // degree -> dinv (reuses dinv buffer as counter)
    hipMemsetAsync(dinv, 0, (size_t)N * sizeof(float), stream);
    deg_count<<<(E + THREADS - 1) / THREADS, THREADS, 0, stream>>>(dst, dinv, E);
    deg_to_dinv<<<(N + THREADS - 1) / THREADS, THREADS, 0, stream>>>(dinv, N);

    // ----- layer 1 -----
    wmma_gemm<4><<<(ntilesN + 3) / 4, 128, 0, stream>>>(x, W1, hw, N, ntilesN);
    hipMemsetAsync(agg, 0, (size_t)N * 32 * sizeof(float), stream);
    scatter_agg<<<(unsigned)((EC + THREADS - 1) / THREADS), THREADS, 0, stream>>>(
        src, dst, dinv, hw, agg, E);
    finalize_layer<<<(unsigned)((NC + THREADS - 1) / THREADS), THREADS, 0, stream>>>(
        agg, hw, dinv, b1, N);                               // h1 in agg

    // ----- layer 2 -----
    wmma_gemm<1><<<(ntilesN + 3) / 4, 128, 0, stream>>>(agg, W2, hw, N, ntilesN);
    hipMemsetAsync(agg, 0, (size_t)N * 32 * sizeof(float), stream);
    scatter_agg<<<(unsigned)((EC + THREADS - 1) / THREADS), THREADS, 0, stream>>>(
        src, dst, dinv, hw, agg, E);
    finalize_layer<<<(unsigned)((NC + THREADS - 1) / THREADS), THREADS, 0, stream>>>(
        agg, hw, dinv, b2, N);                               // h2 in agg

    // ----- heads -----
    edge_head<<<(ntilesE + 3) / 4, 128, 0, stream>>>(
        agg, tfi, gni, ef, eW1, eb1, eW2, eb2, outp, EP, ntilesE);
    gene_head<<<(ntilesG + 3) / 4, 128, 0, stream>>>(
        agg, gid, gW1, gb1, gW2, gb2, outp + EP, NG, ntilesG);
}
